// BinaryConv2d_10325101380263
// MI455X (gfx1250) — matrix operand order repaired and checked by reading, not compiled, not run on hardware
//
#include <hip/hip_runtime.h>
#include <hip/hip_bf16.h>

typedef __attribute__((ext_vector_type(16))) __bf16 v16bf;
typedef __attribute__((ext_vector_type(8)))  __bf16 v8bf;
typedef __attribute__((ext_vector_type(8)))  float  v8f;
typedef int v4i __attribute__((vector_size(16)));   // GCC-style, matches builtin proto

#define GLOBAL_AS __attribute__((address_space(1)))
#define LDS_AS    __attribute__((address_space(3)))

#define C_IN   128
#define C_OUTC 256
#define HH     56
#define WWID   56
#define KDIM   1152            // C_IN * 9
#define XP_W   128             // padded NHWC: ci stride 1, w stride 128
#define XP_H   (58 * 128)      // 7424
#define XP_N   (58 * 7424)     // 430592
#define XP_TOTAL (32 * 430592) // 13778944 elements per (hi|lo) array
#define SPATIAL 3136           // 56*56
#define NTOT   (32 * 3136)     // 100352

// workspace layout (bytes)
#define WB_OFF  0u                         // 256*1152*2 = 589824 B
#define XHI_OFF (1u << 20)                 // 1 MB
#define XLO_OFF (XHI_OFF + XP_TOTAL * 2u)  // + 27.56 MB

// LDS tile geometry: 128 columns x 128 ci, pitch 136 (272B) -> conflict-free banks
#define LDS_COL   136
#define LDS_PLANE (128 * LDS_COL)  // 17408 elements (hi or lo plane)
#define LDS_BUF   (2 * LDS_PLANE)  // 34816 elements per buffer
// total: 2 buffers * 34816 * 2B = 139264 B of LDS

#if __has_builtin(__builtin_amdgcn_global_load_async_to_lds_b128)
#define HAVE_ASYNC_LDS 1
#else
#define HAVE_ASYNC_LDS 0
#endif

// ---------------- pre-pass: binarize + repack weights to [co][p][ci] bf16 ----
__global__ void prep_w(const float* __restrict__ w, __bf16* __restrict__ wb) {
    int t  = blockIdx.x * 256 + threadIdx.x;      // t < 256*1152
    int co = t / KDIM;
    int r  = t - co * KDIM;
    int p  = r >> 7;          // (kh*3+kw) in 0..8
    int ci = r & 127;
    float v = w[co * KDIM + ci * 9 + p];          // OIHW source
    float s = (v > 0.f) ? 1.f : ((v < 0.f) ? -1.f : 0.f);
    wb[t] = (__bf16)s;                            // exact in bf16
}

// ------------- pre-pass: x -> padded NHWC, split into bf16 hi + lo ----------
__global__ void prep_x(const float* __restrict__ x,
                       __bf16* __restrict__ xh, __bf16* __restrict__ xl) {
    int t = blockIdx.x * 256 + threadIdx.x;
    if (t >= XP_TOTAL) return;
    int ci = t & 127;
    int r  = t >> 7;
    int wp = r % 58;
    int r2 = r / 58;
    int hp = r2 % 58;
    int nb = r2 / 58;
    float v = 0.f;
    if (hp >= 1 && hp <= HH && wp >= 1 && wp <= WWID) {
        v = x[((nb * C_IN + ci) * HH + (hp - 1)) * WWID + (wp - 1)];
    }
    __bf16 hi = (__bf16)v;
    xh[t] = hi;
    xl[t] = (__bf16)(v - (float)hi);              // residual: hi+lo ~ fp32
}

// ---------------- main implicit-GEMM conv via bf16 WMMA + async LDS --------
// grid: (NTOT/128, C_OUT/128), block: 256 threads = 8 waves (2 M x 4 N)
// wave tile: 64(M) x 32(N) = 4 mi x 2 ni accumulators
__launch_bounds__(256)
__global__ void bconv_wmma(const __bf16* __restrict__ wb,
                           const __bf16* __restrict__ xh,
                           const __bf16* __restrict__ xl,
                           const float* __restrict__ bias,
                           float* __restrict__ out) {
    __shared__ __bf16 smem[2 * LDS_BUF];          // 139264 B, double-buffered

    const int lane   = threadIdx.x & 31;
    const int wid    = threadIdx.x >> 5;
    const int wave_m = wid & 1;                 // 0..1 (M), 64 rows each
    const int wave_n = wid >> 1;                // 0..3 (N), 32 cols each
    const int half   = lane >> 4;
    const int lrow   = lane & 15;

    const int co_base = blockIdx.y * 128 + wave_m * 64;
    const int s_base  = blockIdx.x * 128;

    // ---- staging: 256 threads move 128 cols x 2 planes, 256B (full col) each
    const int tcol   = threadIdx.x & 127;
    const int tplane = threadIdx.x >> 7;        // 0=hi, 1=lo
    int ts   = s_base + tcol;
    int tnb  = ts / SPATIAL;
    int trem = ts - tnb * SPATIAL;
    int tho  = trem / WWID;
    int two  = trem - tho * WWID;
    const __bf16* tsrc =
        (tplane ? xl : xh) + tnb * XP_N + tho * XP_H + two * XP_W;
    __bf16* tdst0 = &smem[tplane * LDS_PLANE + tcol * LDS_COL];

    auto stage = [&](int buf, int pofs) {
        const __bf16* src = tsrc + pofs;
        __bf16* dst = tdst0 + buf * LDS_BUF;
#if HAVE_ASYNC_LDS
#pragma unroll
        for (int i = 0; i < 16; ++i) {
            __builtin_amdgcn_global_load_async_to_lds_b128(
                (GLOBAL_AS v4i*)(src + i * 8), (LDS_AS v4i*)(dst + i * 8), 0, 0);
        }
#else
#pragma unroll
        for (int i = 0; i < 16; ++i) {
            *(v8bf*)(dst + i * 8) = *(const v8bf*)(src + i * 8);
        }
#endif
    };

    // A-fragment row pointers (per lane K = half*8+[0..7], 16+half*8+[0..7])
    const __bf16* aptr[4];
#pragma unroll
    for (int mi = 0; mi < 4; ++mi)
        aptr[mi] = wb + (co_base + mi * 16 + lrow) * KDIM;

    // B-column spatial decomposition for epilogue indexing + LDS offsets
    int snb[2], srem[2], nfo[2];
#pragma unroll
    for (int ni = 0; ni < 2; ++ni) {
        int col = wave_n * 32 + ni * 16 + lrow;
        int s   = s_base + col;
        int nb  = s / SPATIAL;
        int rem = s - nb * SPATIAL;
        snb[ni]  = nb;
        srem[ni] = rem;
        nfo[ni]  = col * LDS_COL + half * 8;
    }

    v8f acc[4][2] = {};

    stage(0, 0);                                  // prologue: tap p=0 -> buf0

#pragma unroll
    for (int p = 0; p < 9; ++p) {
        if (p + 1 < 9) {
            const int pn = p + 1;
            stage(pn & 1, (pn / 3) * XP_H + (pn % 3) * XP_W);
        }
#if HAVE_ASYNC_LDS
        if (p + 1 < 9) asm volatile("s_wait_asynccnt 0x10" ::: "memory");
        else           asm volatile("s_wait_asynccnt 0x0" ::: "memory");
#endif
        __syncthreads();                          // current buffer ready

        const __bf16* bufh = &smem[(p & 1) * LDS_BUF];
        const __bf16* bufl = bufh + LDS_PLANE;

#pragma unroll
        for (int c = 0; c < 4; ++c) {
            const int k0 = p * 128 + c * 32;
            v16bf afr[4];
#pragma unroll
            for (int mi = 0; mi < 4; ++mi) {
                const __bf16* ap = aptr[mi] + k0 + half * 8;
                union { v16bf v; v8bf h[2]; } u;
                u.h[0] = *(const v8bf*)(ap);
                u.h[1] = *(const v8bf*)(ap + 16);
                afr[mi] = u.v;
            }
            v16bf bfh[2], bfl[2];
#pragma unroll
            for (int ni = 0; ni < 2; ++ni) {
                union { v16bf v; v8bf h[2]; } uh, ul;
                uh.h[0] = *(const v8bf*)(bufh + nfo[ni] + c * 32);
                uh.h[1] = *(const v8bf*)(bufh + nfo[ni] + c * 32 + 16);
                ul.h[0] = *(const v8bf*)(bufl + nfo[ni] + c * 32);
                ul.h[1] = *(const v8bf*)(bufl + nfo[ni] + c * 32 + 16);
                bfh[ni] = uh.v;
                bfl[ni] = ul.v;
            }
#pragma unroll
            for (int mi = 0; mi < 4; ++mi) {
#pragma unroll
                for (int ni = 0; ni < 2; ++ni) {
                    acc[mi][ni] = __builtin_amdgcn_wmma_f32_16x16x32_bf16(
                        false, afr[mi], false, bfh[ni], (short)0,
                        acc[mi][ni], false, false);
                    acc[mi][ni] = __builtin_amdgcn_wmma_f32_16x16x32_bf16(
                        false, afr[mi], false, bfl[ni], (short)0,
                        acc[mi][ni], false, false);
                }
            }
        }
        __syncthreads();   // all waves done reading before buffer is re-staged
    }

    // ---- epilogue: C/D layout lane(0-15): M=r, lane(16-31): M=8+r; N=lrow ----
#pragma unroll
    for (int mi = 0; mi < 4; ++mi) {
        float bv[8];
#pragma unroll
        for (int r = 0; r < 8; ++r)
            bv[r] = bias[co_base + mi * 16 + half * 8 + r];
#pragma unroll
        for (int ni = 0; ni < 2; ++ni) {
            const int obase = snb[ni] * (C_OUTC * SPATIAL) + srem[ni];
#pragma unroll
            for (int r = 0; r < 8; ++r) {
                int co = co_base + mi * 16 + half * 8 + r;
                out[obase + co * SPATIAL] = acc[mi][ni][r] + bv[r];
            }
        }
    }
}

extern "C" void kernel_launch(void* const* d_in, const int* in_sizes, int n_in,
                              void* d_out, int out_size, void* d_ws, size_t ws_size,
                              hipStream_t stream) {
    (void)in_sizes; (void)n_in; (void)out_size; (void)ws_size;
    const float* x    = (const float*)d_in[0];
    const float* w    = (const float*)d_in[1];
    const float* bias = (const float*)d_in[2];
    float* out = (float*)d_out;

    __bf16* wb = (__bf16*)((char*)d_ws + WB_OFF);
    __bf16* xh = (__bf16*)((char*)d_ws + XHI_OFF);
    __bf16* xl = (__bf16*)((char*)d_ws + XLO_OFF);

    prep_w<<<(C_OUTC * KDIM) / 256, 256, 0, stream>>>(w, wb);
    prep_x<<<(XP_TOTAL + 255) / 256, 256, 0, stream>>>(x, xh, xl);

    dim3 grid(NTOT / 128, C_OUTC / 128);
    bconv_wmma<<<grid, 256, 0, stream>>>(wb, xh, xl, bias, out);
}
